// DeltaNetCell_53687091200683
// MI455X (gfx1250) — matrix operand order repaired and checked by reading, not compiled, not run on hardware
//
#include <hip/hip_runtime.h>
#include <math.h>

// ---------------- problem constants ----------------
#define B_    8
#define S_    8192
#define DIN_  512
#define DH_   512
#define HN_   4
#define HD_   128          // DH/HN
#define NTOK  (B_*S_)      // 65536
#define NC_   64           // scan chunks per sequence
#define LCH_  (S_/NC_)     // 128 steps per chunk (multiple of ref CHUNK=16)
#define PF_   16           // async-prefetch granule (== reference CHUNK)
#define LNEPS 1e-5f
#define CLAMP_ 1e-6f

typedef __bf16 bf16;
typedef __bf16 v16bf __attribute__((ext_vector_type(16)));
typedef __bf16 v8bf  __attribute__((ext_vector_type(8)));
typedef __bf16 v4bf  __attribute__((ext_vector_type(4)));
typedef float  v8f   __attribute__((ext_vector_type(8)));

static __device__ __forceinline__ float sigmoidf_(float x) {
    return 1.0f / (1.0f + expf(-x));
}

// =====================================================================
// Kernel 0: one-shot fp32 -> bf16 weight conversion (W is L2-resident,
// so GEMM B-fragments become pure v16bf loads with zero VALU converts)
// =====================================================================
__global__ __launch_bounds__(256)
void k_cvt_w(const float* __restrict__ src, bf16* __restrict__ dst) {
    const int i = (blockIdx.x * 256 + threadIdx.x) * 4;
    float4 v = *(const float4*)(src + i);
    v4bf o;
    o[0] = (bf16)v.x; o[1] = (bf16)v.y; o[2] = (bf16)v.z; o[3] = (bf16)v.w;
    *(v4bf*)(dst + i) = o;
}

// =====================================================================
// Kernel 1: xp = x @ W_in^T  (bf16 WMMA, f32 acc)  + LayerNorm + gates
//           -> a (f32, (NTOK,HN)) and b = beta*xn (bf16, (NTOK,DH))
// Block: 256 threads (8 waves), owns 16 tokens x full 512 cols.
// =====================================================================
__global__ __launch_bounds__(256)
void k_gemm_in_ln_gates(const float* __restrict__ x,       // (NTOK, DIN)
                        const bf16*  __restrict__ W_in,    // (DH, DIN) bf16
                        const float* __restrict__ gamma,   // (DH)
                        const float* __restrict__ lnbeta,  // (DH)
                        const float* __restrict__ W_gate,  // (8, DH)
                        const float* __restrict__ b_gate,  // (8)
                        const float* __restrict__ eig_raw, // (HN)
                        float* __restrict__ a_out,         // (NTOK, HN)
                        bf16*  __restrict__ b_out)         // (NTOK, DH)
{
    __shared__ float xp[16 * DH_];       // 32 KB row-block of xp / xn
    __shared__ float red[16 * 16];
    __shared__ float red2[16 * 16];
    __shared__ float gred[16][16][8];    // gate partials
    __shared__ float tokstat[16][2];     // mean, rstd
    __shared__ float tokab[16][8];       // a(4), beta(4)

    const int tid  = threadIdx.x;
    const int lane = tid & 31;
    const int wave = tid >> 5;
    const int mrow = lane & 15;
    const int hi   = lane >> 4;
    const long m0  = (long)blockIdx.x * 16;

    v8f acc[4];
    #pragma unroll
    for (int t = 0; t < 4; ++t)
        #pragma unroll
        for (int r = 0; r < 8; ++r) acc[t][r] = 0.0f;

    const float* xrow = x + (m0 + mrow) * (long)DIN_;

    for (int ks = 0; ks < DIN_ / 32; ++ks) {
        const int k0 = ks * 32;
        __builtin_prefetch(xrow + k0 + 64, 0, 1);   // global_prefetch_b8
        // ---- A fragment: lane(m=mrow), hi half -> K {hi*16..+15}
        float4 a0 = *(const float4*)(xrow + k0 + hi * 16);
        float4 a1 = *(const float4*)(xrow + k0 + hi * 16 + 4);
        float4 a2 = *(const float4*)(xrow + k0 + hi * 16 + 8);
        float4 a3 = *(const float4*)(xrow + k0 + hi * 16 + 12);
        v16bf af;
        af[0]=(bf16)a0.x; af[1]=(bf16)a0.y; af[2]=(bf16)a0.z; af[3]=(bf16)a0.w;
        af[4]=(bf16)a1.x; af[5]=(bf16)a1.y; af[6]=(bf16)a1.z; af[7]=(bf16)a1.w;
        af[8]=(bf16)a2.x; af[9]=(bf16)a2.y; af[10]=(bf16)a2.z; af[11]=(bf16)a2.w;
        af[12]=(bf16)a3.x; af[13]=(bf16)a3.y; af[14]=(bf16)a3.z; af[15]=(bf16)a3.w;

        #pragma unroll
        for (int t = 0; t < 4; ++t) {
            const int n = (wave * 4 + t) * 16 + mrow;    // B col = lane&15
            // bf16 B fragment: 16 contiguous bf16 (32B aligned) -> 2x b128 loads
            v16bf bf_ = *(const v16bf*)(W_in + (long)n * DIN_ + k0 + hi * 16);
            acc[t] = __builtin_amdgcn_wmma_f32_16x16x32_bf16(
                false, af, false, bf_, (short)0, acc[t], false, false);
        }
    }

    // ---- spill D frags to LDS: reg r holds row (r + 8*hi), col lane&15
    #pragma unroll
    for (int t = 0; t < 4; ++t)
        #pragma unroll
        for (int r = 0; r < 8; ++r)
            xp[(r + 8 * hi) * DH_ + (wave * 4 + t) * 16 + mrow] = acc[t][r];
    __syncthreads();

    // ---- LayerNorm: 16 threads per token
    const int tk = tid >> 4;
    const int th = tid & 15;
    {
        float s = 0.0f, s2 = 0.0f;
        #pragma unroll 8
        for (int j = 0; j < 32; ++j) {
            float v = xp[tk * DH_ + th + 16 * j];
            s += v; s2 += v * v;
        }
        red[tk * 16 + th] = s;
        red2[tk * 16 + th] = s2;
    }
    __syncthreads();
    if (th == 0) {
        float ts = 0.0f, ts2 = 0.0f;
        #pragma unroll
        for (int j = 0; j < 16; ++j) { ts += red[tk * 16 + j]; ts2 += red2[tk * 16 + j]; }
        float mu  = ts * (1.0f / DH_);
        float var = ts2 * (1.0f / DH_) - mu * mu;     // biased var, as torch LN
        tokstat[tk][0] = mu;
        tokstat[tk][1] = rsqrtf(var + LNEPS);
    }
    __syncthreads();
    {
        const float mu = tokstat[tk][0], rstd = tokstat[tk][1];
        #pragma unroll 8
        for (int j = 0; j < 32; ++j) {
            const int c = th + 16 * j;
            float v = xp[tk * DH_ + c];
            xp[tk * DH_ + c] = (v - mu) * rstd * gamma[c] + lnbeta[c];
        }
    }
    __syncthreads();

    // ---- gates: g = xn . W_gate[g] + b_gate[g], 8 gates
    {
        float gp[8];
        #pragma unroll
        for (int g = 0; g < 8; ++g) gp[g] = 0.0f;
        for (int j = 0; j < 32; ++j) {
            const int c = th + 16 * j;
            const float v = xp[tk * DH_ + c];
            #pragma unroll
            for (int g = 0; g < 8; ++g) gp[g] += v * W_gate[g * DH_ + c];
        }
        #pragma unroll
        for (int g = 0; g < 8; ++g) gred[tk][th][g] = gp[g];
    }
    __syncthreads();
    if (th < 8) {
        float s = b_gate[th];
        #pragma unroll
        for (int j = 0; j < 16; ++j) s += gred[tk][j][th];
        const float sg = sigmoidf_(s);
        if (th < HN_) tokab[tk][th] = tanhf(eig_raw[th]) * sg;  // a = eig*alpha
        else          tokab[tk][th] = sg;                        // beta
    }
    __syncthreads();

    if (th < HN_) a_out[(m0 + tk) * HN_ + th] = tokab[tk][th];
    #pragma unroll 8
    for (int j = 0; j < 32; ++j) {
        const int c = th + 16 * j;
        const int hn = c >> 7;
        const float bv = tokab[tk][4 + hn] * xp[tk * DH_ + c];
        b_out[(m0 + tk) * DH_ + c] = (bf16)bv;
    }
}

// =====================================================================
// Kernel 2: phase A of chunked scan. One block per (b, hn, chunk),
// 128 threads (4 waves) = HD channels. b-slab is consumed strictly
// sequentially -> double-buffer 16-step granules into LDS with the
// CDNA5 async-to-LDS engine (ASYNCcnt + barrier gates consumption).
// 16-step granule == reference CHUNK, so the clamped-cumprod window
// semantics are reproduced exactly; carry dependence stays linear in
// the raw cumprod.
// =====================================================================
__global__ __launch_bounds__(HD_)
void k_scan_chunks(const float* __restrict__ a,     // (NTOK, HN)
                   const bf16*  __restrict__ bv,    // (NTOK, DH)
                   bf16*  __restrict__ hloc,        // (NTOK, DH) local scan
                   float* __restrict__ cumA,        // (NTOK, HN) raw cumprod
                   float* __restrict__ hend)        // (B*HN*NC, HD)
{
    __shared__ bf16 sbuf[2][PF_ * HD_];   // 2 x 4 KB granule buffers

    const int blk = blockIdx.x;          // bh*NC + c
    const int c   = blk % NC_;
    const int bh  = blk / NC_;
    const int hn  = bh % HN_;
    const int bb  = bh / HN_;
    const int ch  = threadIdx.x;
    const long t0 = (long)bb * S_ + (long)c * LCH_;

    // per-lane async-copy coordinates: 2 x b128 (16B = 8 channels) per granule
    const int toff0 = ch >> 4;           // steps 0..7   within granule
    const int toff1 = (ch >> 4) + 8;     // steps 8..15  within granule
    const int c8    = (ch & 15) * 8;     // 8-channel group

    typedef __attribute__((address_space(3))) bf16 lds_bf16;

    #define PREFETCH_GRANULE(G, SEL)                                           \
        do {                                                                   \
            const bf16* gp0 = bv + ((t0 + (G) * PF_ + toff0) * DH_ +           \
                                    hn * HD_ + c8);                            \
            const bf16* gp1 = bv + ((t0 + (G) * PF_ + toff1) * DH_ +           \
                                    hn * HD_ + c8);                            \
            unsigned l0 = (unsigned)(unsigned long)                            \
                (lds_bf16*)(&sbuf[(SEL)][toff0 * HD_ + c8]);                   \
            unsigned l1 = (unsigned)(unsigned long)                            \
                (lds_bf16*)(&sbuf[(SEL)][toff1 * HD_ + c8]);                   \
            asm volatile("global_load_async_to_lds_b128 %0, %1, off"           \
                         :: "v"(l0), "v"(gp0) : "memory");                     \
            asm volatile("global_load_async_to_lds_b128 %0, %1, off"           \
                         :: "v"(l1), "v"(gp1) : "memory");                     \
        } while (0)

    PREFETCH_GRANULE(0, 0);

    float h = 0.0f, cum = 1.0f;
    float hbase = 0.0f, wcum = 1.0f, ssum = 0.0f;
    int sel = 0;
    for (int g = 0; g < LCH_ / PF_; ++g) {
        if (g + 1 < LCH_ / PF_) {
            PREFETCH_GRANULE(g + 1, sel ^ 1);
            asm volatile("s_wait_asynccnt 0x2" ::: "memory");  // granule g done
        } else {
            asm volatile("s_wait_asynccnt 0x0" ::: "memory");
        }
        __syncthreads();                   // cross-wave visibility of LDS slab
        #pragma unroll 4
        for (int i2 = 0; i2 < PF_; ++i2) {
            const long t = t0 + g * PF_ + i2;
            const float av = a[t * HN_ + hn];
            const float b  = (float)sbuf[sel][i2 * HD_ + ch];
            if (i2 == 0) { hbase = h; wcum = 1.0f; ssum = 0.0f; }  // CHUNK reset
            wcum *= av;
            const float wc = fmaxf(wcum, CLAMP_);
            ssum += b / wc;
            h = wcum * hbase + wc * ssum;
            cum *= av;
            hloc[t * DH_ + hn * HD_ + ch] = (bf16)h;
            if (ch == 0) cumA[t * HN_ + hn] = cum;
        }
        __syncthreads();                   // everyone done before buffer reuse
        sel ^= 1;
    }
    hend[(long)blk * HD_ + ch] = h;
    #undef PREFETCH_GRANULE
}

// =====================================================================
// Kernel 3: phase B — serial scan over chunk carries per (b,hn) channel.
// carry_0 = h0; carry_{c+1} = hend_c + cumA_end_c * carry_c
// =====================================================================
__global__ __launch_bounds__(HD_)
void k_scan_carry(const float* __restrict__ h0,    // (B, DH)
                  const float* __restrict__ hend,  // (B*HN*NC, HD)
                  const float* __restrict__ cumA,  // (NTOK, HN)
                  float* __restrict__ carry)       // (B*HN*NC, HD)
{
    const int bh = blockIdx.x;
    const int hn = bh % HN_;
    const int bb = bh / HN_;
    const int ch = threadIdx.x;
    float cv = h0[bb * DH_ + hn * HD_ + ch];
    for (int c = 0; c < NC_; ++c) {
        const long idx = (long)bh * NC_ + c;
        carry[idx * HD_ + ch] = cv;
        const long tend = (long)bb * S_ + c * LCH_ + (LCH_ - 1);
        cv = hend[idx * HD_ + ch] + cumA[tend * HN_ + hn] * cv;
    }
}

// =====================================================================
// Kernel 4: phase C — fix-up: h = hloc + cumA * carry; emit bf16 h_all
// for the output GEMM and the fp32 final hidden state.
// =====================================================================
__global__ __launch_bounds__(256)
void k_fixup(const bf16*  __restrict__ hloc,   // (NTOK, DH)
             const float* __restrict__ cumA,   // (NTOK, HN)
             const float* __restrict__ carry,  // (B*HN*NC, HD)
             bf16*  __restrict__ hall,         // (NTOK, DH)
             float* __restrict__ hlast)        // (B, DH)
{
    const int blk = blockIdx.x;        // bb*NC + c
    const int c   = blk % NC_;
    const int bb  = blk / NC_;
    for (int e = threadIdx.x; e < LCH_ * DH_; e += 256) {
        const int i   = e / DH_;
        const int chf = e % DH_;
        const int hn  = chf >> 7;
        const int ch  = chf & (HD_ - 1);
        const long t  = (long)bb * S_ + c * LCH_ + i;
        const float cr = carry[(((long)(bb * HN_ + hn)) * NC_ + c) * HD_ + ch];
        const float h  = (float)hloc[t * DH_ + chf] + cumA[t * HN_ + hn] * cr;
        hall[t * DH_ + chf] = (bf16)h;
        if (c == NC_ - 1 && i == LCH_ - 1) hlast[bb * DH_ + chf] = h;
    }
}

// =====================================================================
// Kernel 5: out = h_all @ W_out^T  (bf16 WMMA, f32 acc) -> fp32 d_out
// =====================================================================
__global__ __launch_bounds__(256)
void k_gemm_out(const bf16*  __restrict__ hall,   // (NTOK, DH) bf16
                const bf16*  __restrict__ W_out,  // (DH, DH) bf16
                float* __restrict__ out)          // (NTOK, DH)
{
    const int tid  = threadIdx.x;
    const int lane = tid & 31;
    const int wave = tid >> 5;
    const int mrow = lane & 15;
    const int hi   = lane >> 4;
    const long m0  = (long)blockIdx.x * 16;

    v8f acc[4];
    #pragma unroll
    for (int t = 0; t < 4; ++t)
        #pragma unroll
        for (int r = 0; r < 8; ++r) acc[t][r] = 0.0f;

    const bf16* hrow = hall + (m0 + mrow) * (long)DH_;

    for (int ks = 0; ks < DH_ / 32; ++ks) {
        const int k0 = ks * 32;
        __builtin_prefetch(hrow + k0 + 64, 0, 1);
        v8bf lo  = *(const v8bf*)(hrow + k0 + hi * 16);
        v8bf hi8 = *(const v8bf*)(hrow + k0 + hi * 16 + 8);
        v16bf af;
        #pragma unroll
        for (int j = 0; j < 8; ++j) { af[j] = lo[j]; af[8 + j] = hi8[j]; }

        #pragma unroll
        for (int t = 0; t < 4; ++t) {
            const int n = (wave * 4 + t) * 16 + mrow;
            v16bf bf_ = *(const v16bf*)(W_out + (long)n * DH_ + k0 + hi * 16);
            acc[t] = __builtin_amdgcn_wmma_f32_16x16x32_bf16(
                false, af, false, bf_, (short)0, acc[t], false, false);
        }
    }

    #pragma unroll
    for (int t = 0; t < 4; ++t)
        #pragma unroll
        for (int r = 0; r < 8; ++r)
            out[(m0 + r + 8 * hi) * (long)DH_ + (wave * 4 + t) * 16 + mrow] = acc[t][r];
}

// =====================================================================
// Host launcher
// =====================================================================
extern "C" void kernel_launch(void* const* d_in, const int* in_sizes, int n_in,
                              void* d_out, int out_size, void* d_ws, size_t ws_size,
                              hipStream_t stream) {
    const float* x       = (const float*)d_in[0];   // (B,S,DIN)
    const float* h0      = (const float*)d_in[1];   // (B,DH)
    const float* W_in    = (const float*)d_in[2];   // (DH,DIN)
    const float* gamma   = (const float*)d_in[3];   // (DH)
    const float* lnbeta  = (const float*)d_in[4];   // (DH)
    const float* W_gate  = (const float*)d_in[5];   // (8,DH)
    const float* b_gate  = (const float*)d_in[6];   // (8)
    const float* eig_raw = (const float*)d_in[7];   // (HN)
    const float* W_out   = (const float*)d_in[8];   // (DH,DH)

    float* out   = (float*)d_out;                         // (B,S,DH)
    float* hlast = (float*)d_out + (size_t)NTOK * DH_;    // (B,DH)

    // workspace carve-up (~197 MiB)
    char* ws = (char*)d_ws;
    size_t off = 0;
    float* a_buf   = (float*)(ws + off); off += (size_t)NTOK * HN_ * sizeof(float);   // 1 MiB
    float* cumA    = (float*)(ws + off); off += (size_t)NTOK * HN_ * sizeof(float);   // 1 MiB
    bf16*  b_buf   = (bf16*) (ws + off); off += (size_t)NTOK * DH_ * sizeof(bf16);    // 64 MiB
    bf16*  hloc    = (bf16*) (ws + off); off += (size_t)NTOK * DH_ * sizeof(bf16);    // 64 MiB
    bf16*  hall    = (bf16*) (ws + off); off += (size_t)NTOK * DH_ * sizeof(bf16);    // 64 MiB
    float* hend    = (float*)(ws + off); off += (size_t)B_ * HN_ * NC_ * HD_ * sizeof(float); // 1 MiB
    float* carry   = (float*)(ws + off); off += (size_t)B_ * HN_ * NC_ * HD_ * sizeof(float); // 1 MiB
    bf16*  Win_bf  = (bf16*) (ws + off); off += (size_t)DH_ * DIN_ * sizeof(bf16);    // 0.5 MiB
    bf16*  Wout_bf = (bf16*) (ws + off); off += (size_t)DH_ * DH_ * sizeof(bf16);     // 0.5 MiB
    (void)ws_size; (void)in_sizes; (void)n_in; (void)out_size;

    // 0) weights -> bf16 (one-shot; L2-resident afterwards)
    k_cvt_w<<<dim3((DH_ * DIN_) / 1024), dim3(256), 0, stream>>>(W_in, Win_bf);
    k_cvt_w<<<dim3((DH_ * DH_) / 1024), dim3(256), 0, stream>>>(W_out, Wout_bf);
    // 1) input GEMM + LN + gates -> a, b
    k_gemm_in_ln_gates<<<dim3(NTOK / 16), dim3(256), 0, stream>>>(
        x, Win_bf, gamma, lnbeta, W_gate, b_gate, eig_raw, a_buf, b_buf);
    // 2) chunk-local scans (async-to-LDS double buffered)
    k_scan_chunks<<<dim3(B_ * HN_ * NC_), dim3(HD_), 0, stream>>>(
        a_buf, b_buf, hloc, cumA, hend);
    // 3) carry scan across chunks
    k_scan_carry<<<dim3(B_ * HN_), dim3(HD_), 0, stream>>>(
        h0, hend, cumA, carry);
    // 4) fix-up -> h_all (bf16) + h_last (f32)
    k_fixup<<<dim3(B_ * NC_), dim3(256), 0, stream>>>(
        hloc, cumA, carry, hall, hlast);
    // 5) output GEMM -> out (f32)
    k_gemm_out<<<dim3(NTOK / 16), dim3(256), 0, stream>>>(hall, Wout_bf, out);
}
